// StyleDecorator_33913061769678
// MI455X (gfx1250) — matrix operand order repaired and checked by reading, not compiled, not run on hardware
//
#include <hip/hip_runtime.h>

// ---------------------------------------------------------------------------
// StyleDecorator for MI455X (gfx1250, wave32, WMMA).
// b=2, C=256, H=W=48, N=H*W=2304, patch=3 -> patch vec len L=2304.
// Pipeline: mean/zm -> cov (WMMA f32) -> Newton-Schulz sqrt/invsqrt (WMMA f32)
//           -> whitening GEMM -> patch matrices (bf16) -> score GEMM
//           (WMMA bf16 16x16x32, 2304^3, async-LDS staged B) -> argmax
//           -> deconv scatter -> coloring GEMM.
// ---------------------------------------------------------------------------

#define CCH   256
#define HW    48
#define NPIX  2304          // 48*48
#define LPAT  2304          // 256*9
#define CC2   65536         // 256*256
#define NS_ITERS 20
#define KC    768           // K-chunk staged in LDS for score GEMM (3 chunks)

typedef __attribute__((ext_vector_type(2)))  float   v2f;
typedef __attribute__((ext_vector_type(8)))  float   v8f;
typedef __attribute__((ext_vector_type(16))) __bf16  v16bf;
typedef __attribute__((ext_vector_type(8)))  __bf16  bf16x8;

union FragBF { v16bf v; bf16x8 h[2]; };

__device__ __forceinline__ unsigned short f2bf(float x) {
  unsigned u = __float_as_uint(x);
  u += 0x7FFFu + ((u >> 16) & 1u);          // round-to-nearest-even
  return (unsigned short)(u >> 16);
}

// ---------------- f32 WMMA 16x16 tile GEMM cores (K step 4) ----------------
// C/D layout: VGPR j -> row m0 + j + 8*(lane>=16), col n0 + (lane&15).
// A 16x4 frag: lane holds A[m0+(lane&15)][k0+2*(lane>>4) + {0,1}]  (b64 load)
// B 4x16 frag (NN): lane holds B[k0+2*(lane>>4)+{0,1}][n0+(lane&15)]

__device__ __forceinline__ v8f gemm_nn(const float* __restrict__ A, int lda,
                                       const float* __restrict__ B, int ldb,
                                       int m0, int n0, int K) {
  const int lane = threadIdx.x & 31;
  const int half = lane >> 4;
  const int l15  = lane & 15;
  const float* ap = A + (size_t)(m0 + l15) * lda + 2 * half;
  const float* bp = B + (size_t)(2 * half) * ldb + n0 + l15;
  v8f acc = {};
  for (int k0 = 0; k0 < K; k0 += 4) {
    v2f a = *(const v2f*)ap;
    v2f b; b.x = bp[0]; b.y = bp[ldb];
    acc = __builtin_amdgcn_wmma_f32_16x16x4_f32(false, a, false, b,
                                                (short)0, acc, false, false);
    ap += 4; bp += 4 * (size_t)ldb;
  }
  return acc;
}

__device__ __forceinline__ v8f gemm_nt(const float* __restrict__ A, int lda,
                                       const float* __restrict__ B, int ldb,
                                       int m0, int n0, int K) {
  const int lane = threadIdx.x & 31;
  const int half = lane >> 4;
  const int l15  = lane & 15;
  const float* ap = A + (size_t)(m0 + l15) * lda + 2 * half;
  const float* bp = B + (size_t)(n0 + l15) * ldb + 2 * half;
  v8f acc = {};
  for (int k0 = 0; k0 < K; k0 += 4) {
    v2f a = *(const v2f*)ap;
    v2f b = *(const v2f*)bp;
    acc = __builtin_amdgcn_wmma_f32_16x16x4_f32(false, a, false, b,
                                                (short)0, acc, false, false);
    ap += 4; bp += 4;
  }
  return acc;
}

template <class F>
__device__ __forceinline__ void store16(float* __restrict__ C, int ldc,
                                        int m0, int n0, const v8f& acc, F f) {
  const int lane = threadIdx.x & 31;
  const int half = lane >> 4;
  const int col  = n0 + (lane & 15);
#pragma unroll
  for (int j = 0; j < 8; ++j) {
    int row = m0 + j + 8 * half;
    C[(size_t)row * ldc + col] = f(acc[j], row, col);
  }
}

// ------------------------------- kernels -----------------------------------

// grid(256 channels, 2 tensors), block 256.  zm = x - mean ; record means.
__global__ void sd_meanzm(const float* __restrict__ cin,
                          const float* __restrict__ sin,
                          float* __restrict__ zm, float* __restrict__ means) {
  __shared__ float red[256];
  const int ch = blockIdx.x, t = blockIdx.y, tid = threadIdx.x;
  const float* src = (t == 0 ? cin : sin) + (size_t)ch * NPIX;
  float s = 0.f;
  for (int i = tid; i < NPIX; i += 256) s += src[i];
  red[tid] = s; __syncthreads();
  for (int off = 128; off; off >>= 1) {
    if (tid < off) red[tid] += red[tid + off];
    __syncthreads();
  }
  const float mean = red[0] / (float)NPIX;
  if (tid == 0) means[t * CCH + ch] = mean;
  float* dst = zm + (size_t)t * CCH * NPIX + (size_t)ch * NPIX;
  for (int i = tid; i < NPIX; i += 256) dst[i] = src[i] - mean;
}

// cov = zm * zm^T (unnormalized, matches torch).  grid(2,16,2), block 256.
__global__ void sd_cov(const float* __restrict__ zm, float* __restrict__ cov) {
  const int t = blockIdx.z;
  const float* A = zm + (size_t)t * CCH * NPIX;
  const int w  = threadIdx.x >> 5;
  const int m0 = blockIdx.x * 128 + w * 16;
  const int n0 = blockIdx.y * 16;
  v8f acc = gemm_nt(A, NPIX, A, NPIX, m0, n0, NPIX);
  store16(cov + (size_t)t * CC2, CCH, m0, n0, acc,
          [](float v, int, int) { return v; });
}

// grid(2 runs), block 256: tr = trace(cov); Y = cov/tr; Z = I.
__global__ void sd_trace_init(const float* __restrict__ cov,
                              float* __restrict__ Y, float* __restrict__ Z,
                              float* __restrict__ tr) {
  __shared__ float red[256];
  const int r = blockIdx.x, tid = threadIdx.x;
  const float* Cv = cov + (size_t)r * CC2;
  red[tid] = Cv[tid * 257];   // diagonal
  __syncthreads();
  for (int off = 128; off; off >>= 1) {
    if (tid < off) red[tid] += red[tid + off];
    __syncthreads();
  }
  if (tid == 0) tr[r] = red[0];
  __syncthreads();
  const float inv = 1.0f / red[0];
  float* Yr = Y + (size_t)r * CC2;
  float* Zr = Z + (size_t)r * CC2;
  for (int i = tid; i < CC2; i += 256) {
    Yr[i] = Cv[i] * inv;
    Zr[i] = ((i >> 8) == (i & 255)) ? 1.0f : 0.0f;
  }
}

// T = 3I - Z*Y.  grid(2,16,2 runs).
__global__ void sd_ns_T(const float* __restrict__ Zc,
                        const float* __restrict__ Yc, float* __restrict__ T) {
  const int r = blockIdx.z;
  const int w  = threadIdx.x >> 5;
  const int m0 = blockIdx.x * 128 + w * 16;
  const int n0 = blockIdx.y * 16;
  v8f acc = gemm_nn(Zc + (size_t)r * CC2, CCH, Yc + (size_t)r * CC2, CCH,
                    m0, n0, CCH);
  store16(T + (size_t)r * CC2, CCH, m0, n0, acc,
          [](float v, int row, int col) {
            return (row == col ? 3.0f : 0.0f) - v;
          });
}

// which=0: Yn = 0.5*Y*T ; which=1: Zn = 0.5*T*Z.  grid(2,16,4).
__global__ void sd_ns_update(const float* __restrict__ Yc,
                             const float* __restrict__ Zc,
                             const float* __restrict__ T,
                             float* __restrict__ Yn, float* __restrict__ Zn) {
  const int r = blockIdx.z & 1, which = blockIdx.z >> 1;
  const float* A = (which == 0) ? Yc + (size_t)r * CC2 : T + (size_t)r * CC2;
  const float* B = (which == 0) ? T  + (size_t)r * CC2 : Zc + (size_t)r * CC2;
  float* O       = (which == 0) ? Yn + (size_t)r * CC2 : Zn + (size_t)r * CC2;
  const int w  = threadIdx.x >> 5;
  const int m0 = blockIdx.x * 128 + w * 16;
  const int n0 = blockIdx.y * 16;
  v8f acc = gemm_nn(A, CCH, B, CCH, m0, n0, CCH);
  store16(O, CCH, m0, n0, acc, [](float v, int, int) { return 0.5f * v; });
}

// norm_t = rsqrt(tr_t) * Z_t * zm_t.  grid(2,144,2 tensors).
__global__ void sd_whiten(const float* __restrict__ Z,
                          const float* __restrict__ zm,
                          const float* __restrict__ tr,
                          float* __restrict__ normf) {
  const int t = blockIdx.z;
  const int w  = threadIdx.x >> 5;
  const int m0 = blockIdx.x * 128 + w * 16;
  const int n0 = blockIdx.y * 16;
  v8f acc = gemm_nn(Z + (size_t)t * CC2, CCH, zm + (size_t)t * CCH * NPIX,
                    NPIX, m0, n0, CCH);
  const float sc = rsqrtf(tr[t]);
  store16(normf + (size_t)t * CCH * NPIX, NPIX, m0, n0, acc,
          [sc](float v, int, int) { return v * sc; });
}

// Build patch matrix (bf16). Row n = flattened (c,ky,kx) 3x3 patch at pixel n.
// If rknorm != nullptr also emit rknorm[n] = rsqrt(||row||^2) (f32 values).
__global__ void sd_patch(const float* __restrict__ nrm,
                         unsigned short* __restrict__ out,
                         float* __restrict__ rknorm) {
  __shared__ float red[256];
  const int n = blockIdx.x, tid = threadIdx.x;
  const int y = n / HW, x = n % HW;
  float ss = 0.f;
  for (int l = tid; l < LPAT; l += 256) {
    const int cch = l / 9, rem = l % 9, ky = rem / 3, kx = rem % 3;
    const int yy = y + ky - 1, xx = x + kx - 1;
    float v = 0.f;
    if (yy >= 0 && yy < HW && xx >= 0 && xx < HW)
      v = nrm[(size_t)cch * NPIX + yy * HW + xx];
    out[(size_t)n * LPAT + l] = f2bf(v);
    ss += v * v;
  }
  if (rknorm != nullptr) {
    red[tid] = ss; __syncthreads();
    for (int off = 128; off; off >>= 1) {
      if (tid < off) red[tid] += red[tid + off];
      __syncthreads();
    }
    if (tid == 0) rknorm[n] = rsqrtf(red[0]);
  }
}

// D[p][n] = Sp[p,:] . Cp[n,:]   (bf16 x bf16 -> f32, 2304^3 GEMM).
// grid(18,144), block 256 = 8 waves, each wave a 16x16 tile of D; all 8 waves
// share the 16-row B tile (Cp rows n0..n0+15), which is staged through LDS in
// KC-column chunks with GLOBAL_LOAD_ASYNC_TO_LDS_B128 (ASYNCcnt) and read back
// as ds_load_b128 fragments.
// bf16 A frag: lane m=m0+(lane&15): chunks [k0+8h .. +7] and [k0+8h+16 .. +7].
// bf16 B frag: lane n=n0+(lane&15): chunks [k0+8h .. +7] and [k0+8h+16 .. +7].
__global__ void sd_score(const unsigned short* __restrict__ Sp_,
                         const unsigned short* __restrict__ Cp_,
                         float* __restrict__ D) {
  __shared__ __align__(16) unsigned short ldsB[16 * KC];   // 24 KB
  const __bf16* Sp = (const __bf16*)Sp_;
  const int w = threadIdx.x >> 5, lane = threadIdx.x & 31;
  const int half = lane >> 4, l15 = lane & 15;
  const int m0 = blockIdx.x * 128 + w * 16;
  const int n0 = blockIdx.y * 16;
  const __bf16* arow = Sp + (size_t)(m0 + l15) * LPAT + 8 * half;
  const __bf16* lbrow = (const __bf16*)ldsB + l15 * KC + 8 * half;
  // base of the 16 Cp rows this block consumes (16B-chunk granularity)
  const unsigned short* gB = Cp_ + (size_t)n0 * LPAT;

  v8f acc = {};
  for (int kc = 0; kc < LPAT; kc += KC) {
    // ---- stage B chunk: 16 rows x KC cols = 16*KC*2 bytes, 16B per lane ----
    for (int i = threadIdx.x; i < (16 * KC) / 8; i += 256) {  // 8 ushort/16B
      const int row = i / (KC / 8), piece = i % (KC / 8);
      const unsigned short* gsrc = gB + (size_t)row * LPAT + kc + piece * 8;
      unsigned lds_off = (unsigned)(size_t)(&ldsB[i * 8]);
      asm volatile("global_load_async_to_lds_b128 %0, %1, off"
                   :: "v"(lds_off), "v"(gsrc) : "memory");
    }
    asm volatile("s_wait_asynccnt 0x0" ::: "memory");
    __syncthreads();

    // ---- consume chunk: KC/32 WMMA steps ----
    for (int k0 = 0; k0 < KC; k0 += 32) {
      FragBF a, b;
      a.h[0] = *(const bf16x8*)(arow + kc + k0);
      a.h[1] = *(const bf16x8*)(arow + kc + k0 + 16);
      b.h[0] = *(const bf16x8*)(lbrow + k0);
      b.h[1] = *(const bf16x8*)(lbrow + k0 + 16);
      acc = __builtin_amdgcn_wmma_f32_16x16x32_bf16(
          false, a.v, false, b.v, (short)0, acc, false, false);
    }
    __syncthreads();   // protect LDS before restaging
  }

  const int col = n0 + l15;
#pragma unroll
  for (int j = 0; j < 8; ++j) {
    int row = m0 + j + 8 * half;
    D[(size_t)row * NPIX + col] = acc[j];
  }
}

// argmax over p of D[p][n] * rknorm[p].  9 blocks x 256.
__global__ void sd_argmax(const float* __restrict__ D,
                          const float* __restrict__ rk,
                          int* __restrict__ idx) {
  const int n = blockIdx.x * 256 + threadIdx.x;
  float best = -3.4e38f; int bi = 0;
  for (int p = 0; p < NPIX; ++p) {
    float v = D[(size_t)p * NPIX + n] * rk[p];
    if (v > best) { best = v; bi = p; }
  }
  idx[n] = bi;
}

// transpose-conv of the one-hot with the style patch kernels + overlap norm.
// out[c,y,x] = sum_{dy,dx} nrm_s[c, py(idx[y+dy,x+dx])-dy, px(...)-dx] (bounded)
__global__ void sd_deconv(const int* __restrict__ idx,
                          const float* __restrict__ nrmS,
                          float* __restrict__ rF) {
  const int gid = blockIdx.x * 256 + threadIdx.x;   // 256*2304 threads
  const int cch = gid / NPIX, n = gid % NPIX;
  const int y = n / HW, x = n % HW;
  float acc = 0.f;
#pragma unroll
  for (int dy = -1; dy <= 1; ++dy) {
#pragma unroll
    for (int dx = -1; dx <= 1; ++dx) {
      const int yy = y + dy, xx = x + dx;
      if (yy >= 0 && yy < HW && xx >= 0 && xx < HW) {
        const int p = idx[yy * HW + xx];
        const int sy = p / HW - dy, sx = p % HW - dx;
        if (sy >= 0 && sy < HW && sx >= 0 && sx < HW)
          acc += nrmS[(size_t)cch * NPIX + sy * HW + sx];
      }
    }
  }
  const float cy = (y == 0 || y == HW - 1) ? 2.f : 3.f;
  const float cx = (x == 0 || x == HW - 1) ? 2.f : 3.f;
  rF[(size_t)cch * NPIX + n] = acc / (cy * cx);
}

// out = sqrt(tr_s) * Y_s * rF + mean_s  (strength==1 -> final output).
// grid(2,144).
__global__ void sd_coloring(const float* __restrict__ Y,
                            const float* __restrict__ rF,
                            const float* __restrict__ tr,
                            const float* __restrict__ means,
                            float* __restrict__ out) {
  const int w  = threadIdx.x >> 5;
  const int m0 = blockIdx.x * 128 + w * 16;
  const int n0 = blockIdx.y * 16;
  v8f acc = gemm_nn(Y + CC2 /*run s*/, CCH, rF, NPIX, m0, n0, CCH);
  const float sc = sqrtf(tr[1]);
  store16(out, NPIX, m0, n0, acc,
          [sc, means](float v, int row, int) {
            return v * sc + means[CCH + row];
          });
}

// --------------------------------- host ------------------------------------

extern "C" void kernel_launch(void* const* d_in, const int* in_sizes, int n_in,
                              void* d_out, int out_size, void* d_ws,
                              size_t ws_size, hipStream_t stream) {
  const float* cin_all = (const float*)d_in[0];
  const float* sin_all = (const float*)d_in[1];
  float* outf = (float*)d_out;

  // carve scratch (reused across batches); total ~58 MB
  char* base = (char*)d_ws;
  auto carve = [&](size_t bytes) -> char* {
    char* p = base;
    base += (bytes + 255) & ~(size_t)255;
    return p;
  };
  float* zm     = (float*)carve((size_t)2 * CCH * NPIX * 4);
  float* means  = (float*)carve((size_t)2 * CCH * 4);
  float* cov    = (float*)carve((size_t)2 * CC2 * 4);
  float* tr     = (float*)carve(64);
  float* nsY    = (float*)carve((size_t)2 * CC2 * 4);
  float* nsZ    = (float*)carve((size_t)2 * CC2 * 4);
  float* nsT    = (float*)carve((size_t)2 * CC2 * 4);
  float* nsY2   = (float*)carve((size_t)2 * CC2 * 4);
  float* nsZ2   = (float*)carve((size_t)2 * CC2 * 4);
  float* normf  = (float*)carve((size_t)2 * CCH * NPIX * 4);
  float* rknorm = (float*)carve((size_t)NPIX * 4);
  float* rF     = (float*)carve((size_t)CCH * NPIX * 4);
  float* score  = (float*)carve((size_t)NPIX * NPIX * 4);
  int*   idxb   = (int*)carve((size_t)NPIX * 4);
  unsigned short* Cp = (unsigned short*)carve((size_t)NPIX * LPAT * 2);
  unsigned short* Sp = (unsigned short*)carve((size_t)NPIX * LPAT * 2);

  const dim3 blk(256);
  for (int b = 0; b < 2; ++b) {
    const float* cin = cin_all + (size_t)b * CCH * NPIX;
    const float* sin = sin_all + (size_t)b * CCH * NPIX;
    float* outb = outf + (size_t)b * CCH * NPIX;

    sd_meanzm<<<dim3(CCH, 2), blk, 0, stream>>>(cin, sin, zm, means);
    sd_cov<<<dim3(2, 16, 2), blk, 0, stream>>>(zm, cov);
    sd_trace_init<<<dim3(2), blk, 0, stream>>>(cov, nsY, nsZ, tr);

    for (int it = 0; it < NS_ITERS; ++it) {
      float* Yc = (it & 1) ? nsY2 : nsY;
      float* Zc = (it & 1) ? nsZ2 : nsZ;
      float* Yn = (it & 1) ? nsY : nsY2;
      float* Zn = (it & 1) ? nsZ : nsZ2;
      sd_ns_T<<<dim3(2, 16, 2), blk, 0, stream>>>(Zc, Yc, nsT);
      sd_ns_update<<<dim3(2, 16, 4), blk, 0, stream>>>(Yc, Zc, nsT, Yn, Zn);
    }
    // NS_ITERS even -> results back in nsY / nsZ.

    sd_whiten<<<dim3(2, 144, 2), blk, 0, stream>>>(nsZ, zm, tr, normf);
    sd_patch<<<dim3(NPIX), blk, 0, stream>>>(normf, Cp, nullptr);
    sd_patch<<<dim3(NPIX), blk, 0, stream>>>(normf + (size_t)CCH * NPIX, Sp,
                                             rknorm);
    sd_score<<<dim3(18, 144), blk, 0, stream>>>(Sp, Cp, score);
    sd_argmax<<<dim3(9), blk, 0, stream>>>(score, rknorm, idxb);
    sd_deconv<<<dim3(NPIX), blk, 0, stream>>>(
        idxb, normf + (size_t)CCH * NPIX, rF);
    sd_coloring<<<dim3(2, 144), blk, 0, stream>>>(nsY, rF, tr, means, outb);
  }
  (void)in_sizes; (void)n_in; (void)out_size; (void)ws_size;
}
